// EMAVectorQuantizer_48155173323477
// MI455X (gfx1250) — matrix operand order repaired and checked by reading, not compiled, not run on hardware
//
#include <hip/hip_runtime.h>

typedef __attribute__((ext_vector_type(8))) float v8f;
typedef __attribute__((ext_vector_type(8))) __bf16 v8bf;
typedef __attribute__((ext_vector_type(16))) __bf16 v16bf;

// Problem constants (from reference)
#define NB 16
#define NT 2048
#define ND 64
#define NK 8192
#define NN (NB * NT)          // 32768 rows
#define DECAY 0.99f
#define ONE_MINUS_DECAY (1.0f - 0.99f)
#define EPS 1e-5f

// LDS row stride in bf16 elements (144B rows: 16B aligned, conflict-broken)
#define WSTRIDE 72
#define CHUNK 64              // codes staged per LDS chunk

// ---- workspace layout ----
// float region:
//   wnorm : [0, 8192)
//   counts: [8192, 16384)
//   dw    : [16384, 16384 + 524288)
//   nsum  : [540672]
// bf16 region (starts at float offset 540688, 32B aligned):
//   whi   : K*D bf16
//   wlo   : K*D bf16
#define WS_WNORM 0
#define WS_COUNTS 8192
#define WS_DW 16384
#define WS_NSUM (16384 + NK * ND)
#define WS_ZERO_LEN (NK + NK * ND + 1)   // counts + dw + nsum
#define WS_BF16_BASE 540688              // float offset, padded

// ---- d_out layout (floats, concatenated in reference return order) ----
#define OUT_Q 0                                   // [N, D]
#define OUT_IDX (NN * ND)                         // [N]
#define OUT_W (OUT_IDX + NN)                      // [K, D]
#define OUT_CS (OUT_W + NK * ND)                  // [K]
#define OUT_EMA (OUT_CS + NK)                     // [K, D]

__device__ __forceinline__ void split_bf16(float f, __bf16& hi, __bf16& lo) {
  hi = (__bf16)f;
  lo = (__bf16)(f - (float)hi);
}

// -------------------------------------------------------------------------
// Init: zero accumulators, split W into bf16 hi/lo, precompute ||w_k||^2
// -------------------------------------------------------------------------
__global__ __launch_bounds__(256) void vq_init_kernel(
    const float* __restrict__ W, float* __restrict__ ws,
    __bf16* __restrict__ whi, __bf16* __restrict__ wlo) {
  int gid = blockIdx.x * blockDim.x + threadIdx.x;
  if (gid < WS_ZERO_LEN) ws[WS_COUNTS + gid] = 0.0f;
  if (gid < NK * ND) {
    float w = W[gid];
    __bf16 h, l;
    split_bf16(w, h, l);
    whi[gid] = h;
    wlo[gid] = l;
  }
  if (gid < NK) {
    const float* wr = W + gid * ND;
    float s = 0.0f;
#pragma unroll
    for (int d = 0; d < ND; ++d) {
      float v = wr[d];
      s += v * v;
    }
    ws[WS_WNORM + gid] = s;
  }
}

// -------------------------------------------------------------------------
// Main: split-bf16 WMMA distance GEMM (fp32-accurate) + fused argmin
// 256 threads = 8 waves; wave w handles rows [blk*128 + w*16, +16)
// -------------------------------------------------------------------------
__global__ __launch_bounds__(256, 2) void vq_argmin_kernel(
    const float* __restrict__ x, const float* __restrict__ W,
    const __bf16* __restrict__ whi, const __bf16* __restrict__ wlo,
    const float* __restrict__ wnorm, float* __restrict__ counts,
    float* __restrict__ dwacc, float* __restrict__ out_q,
    float* __restrict__ out_idx) {
  __shared__ __bf16 lds_hi[CHUNK * WSTRIDE];
  __shared__ __bf16 lds_lo[CHUNK * WSTRIDE];

  const int tid = threadIdx.x;
  const int lane = tid & 31;
  const int wave = tid >> 5;
  const int nl = lane & 15;   // N inside 16-tile / A-row index
  const int h = lane >> 4;    // lane half selects K sub-ranges
  const int row0 = blockIdx.x * 128 + wave * 16;

  // ---- A fragments (bf16 hi/lo), built once, reused for all K chunks.
  // 16-bit A 16x32 layout: elem e(0..7) <-> K = Kb + 8h + e,
  //                        elem e(8..15) <-> K = Kb + 16 + 8h + (e-8)
  v16bf ah[2], al[2];
  {
    const float* xrow = x + (row0 + nl) * ND;
#pragma unroll
    for (int kf = 0; kf < 2; ++kf) {
      const int Kb = 32 * kf;
#pragma unroll
      for (int e = 0; e < 8; ++e) {
        __bf16 hh, ll;
        split_bf16(xrow[Kb + 8 * h + e], hh, ll);
        ah[kf][e] = hh;
        al[kf][e] = ll;
        split_bf16(xrow[Kb + 16 + 8 * h + e], hh, ll);
        ah[kf][e + 8] = hh;
        al[kf][e + 8] = ll;
      }
    }
  }

  float mind[8];
  int midx[8];
#pragma unroll
  for (int r = 0; r < 8; ++r) {
    mind[r] = 3.4e38f;
    midx[r] = 0;
  }

  for (int c0 = 0; c0 < NK; c0 += CHUNK) {
    __syncthreads();
    // cooperative stage of 64-code bf16 hi/lo W chunk into padded LDS
    {
      int r = tid >> 2;
      int q = tid & 3;
      const uint4* sh = reinterpret_cast<const uint4*>(whi + (c0 + r) * ND + q * 16);
      const uint4* sl = reinterpret_cast<const uint4*>(wlo + (c0 + r) * ND + q * 16);
      uint4* dh = reinterpret_cast<uint4*>(&lds_hi[r * WSTRIDE + q * 16]);
      uint4* dl = reinterpret_cast<uint4*>(&lds_lo[r * WSTRIDE + q * 16]);
      dh[0] = sh[0];
      dh[1] = sh[1];
      dl[0] = sl[0];
      dl[1] = sl[1];
      if (c0 + CHUNK < NK)
        __builtin_prefetch(whi + (c0 + CHUNK + r) * ND + q * 16, 0, 0);
    }
    __syncthreads();

#pragma unroll
    for (int sub = 0; sub < 4; ++sub) {
      const int nloc = sub * 16 + nl;
      // bf16 B 32x16 layout: elem e(0..7) <-> K = Kb + 8h + e, e(8..15) <-> K = Kb + 16 + 8h + (e-8), N = lane&15
      const int base = nloc * WSTRIDE + 8 * h;
      v8bf h00 = *(const v8bf*)&lds_hi[base];
      v8bf h01 = *(const v8bf*)&lds_hi[base + 16];
      v8bf h10 = *(const v8bf*)&lds_hi[base + 32];
      v8bf h11 = *(const v8bf*)&lds_hi[base + 48];
      v8bf l00 = *(const v8bf*)&lds_lo[base];
      v8bf l01 = *(const v8bf*)&lds_lo[base + 16];
      v8bf l10 = *(const v8bf*)&lds_lo[base + 32];
      v8bf l11 = *(const v8bf*)&lds_lo[base + 48];
      v16bf bh0 = __builtin_shufflevector(h00, h01, 0, 1, 2, 3, 4, 5, 6, 7, 8,
                                          9, 10, 11, 12, 13, 14, 15);
      v16bf bh1 = __builtin_shufflevector(h10, h11, 0, 1, 2, 3, 4, 5, 6, 7, 8,
                                          9, 10, 11, 12, 13, 14, 15);
      v16bf bl0 = __builtin_shufflevector(l00, l01, 0, 1, 2, 3, 4, 5, 6, 7, 8,
                                          9, 10, 11, 12, 13, 14, 15);
      v16bf bl1 = __builtin_shufflevector(l10, l11, 0, 1, 2, 3, 4, 5, 6, 7, 8,
                                          9, 10, 11, 12, 13, 14, 15);

      // dot = xh*wh + xh*wl + xl*wh  (error-compensated split product)
      v8f acc = {};
      acc = __builtin_amdgcn_wmma_f32_16x16x32_bf16(false, ah[0], false, bh0,
                                                    (short)0, acc, false, false);
      acc = __builtin_amdgcn_wmma_f32_16x16x32_bf16(false, ah[1], false, bh1,
                                                    (short)0, acc, false, false);
      acc = __builtin_amdgcn_wmma_f32_16x16x32_bf16(false, ah[0], false, bl0,
                                                    (short)0, acc, false, false);
      acc = __builtin_amdgcn_wmma_f32_16x16x32_bf16(false, ah[1], false, bl1,
                                                    (short)0, acc, false, false);
      acc = __builtin_amdgcn_wmma_f32_16x16x32_bf16(false, al[0], false, bh0,
                                                    (short)0, acc, false, false);
      acc = __builtin_amdgcn_wmma_f32_16x16x32_bf16(false, al[1], false, bh1,
                                                    (short)0, acc, false, false);

      const int n = c0 + sub * 16 + nl;
      const float wn = wnorm[n];
#pragma unroll
      for (int r = 0; r < 8; ++r) {
        // ||x||^2 dropped: constant per row, argmin-invariant
        float dv = wn - 2.0f * acc[r];
        if (dv < mind[r]) {
          mind[r] = dv;
          midx[r] = n;
        }
      }
    }
  }

  // butterfly (min, argmin) reduction over the 16-lane N groups (wave32)
#pragma unroll
  for (int r = 0; r < 8; ++r) {
    float f = mind[r];
    int i = midx[r];
#pragma unroll
    for (int m = 1; m < 16; m <<= 1) {
      float of = __shfl_xor(f, m, 16);
      int oi = __shfl_xor(i, m, 16);
      if (of < f || (of == f && oi < i)) {
        f = of;
        i = oi;
      }
    }
    mind[r] = f;
    midx[r] = i;
  }
  // rows 0..7 final in lane 0 slots, rows 8..15 final in lane 16 slots

  for (int r = 0; r < 16; ++r) {
    int code = __shfl(midx[r & 7], (r < 8) ? 0 : 16, 32);
    int grow = row0 + r;
    int d = lane;
    const float* wc = W + code * ND;
    // quantized = weight[idx]
    out_q[grow * ND + d] = wc[d];
    out_q[grow * ND + d + 32] = wc[d + 32];
    // dw segment-sum and counts via fp32 global atomics
    float x0 = x[grow * ND + d];
    float x1 = x[grow * ND + d + 32];
    atomicAdd(&dwacc[code * ND + d], x0);
    atomicAdd(&dwacc[code * ND + d + 32], x1);
    if (lane == 0) {
      atomicAdd(&counts[code], 1.0f);
      out_idx[grow] = (float)code;
    }
  }
}

// -------------------------------------------------------------------------
// EMA update: new_cluster_size, new_ema_w, and global n = sum(new_cs)
// -------------------------------------------------------------------------
__global__ __launch_bounds__(256) void vq_ema_kernel(
    const float* __restrict__ cluster_size, const float* __restrict__ ema_w,
    const float* __restrict__ counts, const float* __restrict__ dwacc,
    float* __restrict__ out_cs, float* __restrict__ out_ema,
    float* __restrict__ nsum) {
  int gid = blockIdx.x * blockDim.x + threadIdx.x;  // over K*D
  out_ema[gid] = ema_w[gid] * DECAY + ONE_MINUS_DECAY * dwacc[gid];
  if ((gid & (ND - 1)) == 0) {
    int k = gid / ND;
    float ncs = cluster_size[k] * DECAY + ONE_MINUS_DECAY * counts[k];
    out_cs[k] = ncs;
    atomicAdd(nsum, ncs);
  }
}

// -------------------------------------------------------------------------
// Normalize: new_weight = new_ema_w / ((ncs+eps)/(n+K*eps)*n)
// -------------------------------------------------------------------------
__global__ __launch_bounds__(256) void vq_weight_kernel(
    const float* __restrict__ out_cs, const float* __restrict__ out_ema,
    const float* __restrict__ nsum, float* __restrict__ out_w) {
  int gid = blockIdx.x * blockDim.x + threadIdx.x;  // over K*D
  int k = gid / ND;
  float n = *nsum;
  float ncs = out_cs[k];
  float cs = (ncs + EPS) / (n + (float)NK * EPS) * n;
  out_w[gid] = out_ema[gid] / cs;
}

extern "C" void kernel_launch(void* const* d_in, const int* in_sizes, int n_in,
                              void* d_out, int out_size, void* d_ws,
                              size_t ws_size, hipStream_t stream) {
  const float* x = (const float*)d_in[0];             // [B,T,D]
  const float* weight = (const float*)d_in[1];        // [K,D]
  const float* cluster_size = (const float*)d_in[2];  // [K]
  const float* ema_w = (const float*)d_in[3];         // [K,D]

  float* ws = (float*)d_ws;
  float* wnorm = ws + WS_WNORM;
  float* counts = ws + WS_COUNTS;
  float* dwacc = ws + WS_DW;
  float* nsum = ws + WS_NSUM;
  __bf16* whi = (__bf16*)(ws + WS_BF16_BASE);
  __bf16* wlo = whi + NK * ND;

  float* out = (float*)d_out;
  float* out_q = out + OUT_Q;
  float* out_idx = out + OUT_IDX;
  float* out_w = out + OUT_W;
  float* out_cs = out + OUT_CS;
  float* out_ema = out + OUT_EMA;

  int init_blocks = (WS_ZERO_LEN + 255) / 256;
  vq_init_kernel<<<init_blocks, 256, 0, stream>>>(weight, ws, whi, wlo);

  vq_argmin_kernel<<<NN / 128, 256, 0, stream>>>(x, weight, whi, wlo, wnorm,
                                                 counts, dwacc, out_q, out_idx);

  vq_ema_kernel<<<(NK * ND) / 256, 256, 0, stream>>>(
      cluster_size, ema_w, counts, dwacc, out_cs, out_ema, nsum);

  vq_weight_kernel<<<(NK * ND) / 256, 256, 0, stream>>>(out_cs, out_ema, nsum,
                                                        out_w);
}
